// MultiheadAttention_57226144252585
// MI455X (gfx1250) — compile-verified, matrix operand order checked by reading, not run on hardware
//
#include <hip/hip_runtime.h>
#include <hip/hip_bf16.h>
#include <cstdint>

typedef __attribute__((ext_vector_type(16))) __bf16 v16bf;
typedef __attribute__((ext_vector_type(8)))  __bf16 v8bf;
typedef __attribute__((ext_vector_type(8)))  float  v8f;
typedef __attribute__((ext_vector_type(4)))  float  v4f;

#define NEG_INF (-1e30f)

__device__ __forceinline__ v8f vzero() {
  v8f z;
#pragma unroll
  for (int i = 0; i < 8; ++i) z[i] = 0.f;
  return z;
}

__device__ __forceinline__ v8f wmma_bf16(v16bf a, v16bf b, v8f c) {
  // D = A(16x32) * B(32x16) + C, f32 accumulate
  return __builtin_amdgcn_wmma_f32_16x16x32_bf16(false, a, false, b, (short)0, c,
                                                 false, false);
}

// 16x32 bf16 fragment load (A layout; B uses same pattern on B^T rows):
// lane L<16 : row L, k 0..7 (16B) and k 16..23 ; lane L>=16 : row L-16, k 8..15 / 24..31
__device__ __forceinline__ v16bf load_frag(const __bf16* base, int lane, int strideElems) {
  const __bf16* p = base + (size_t)(lane & 15) * strideElems + ((lane >> 4) & 1) * 8;
  v8bf lo = *(const v8bf*)p;
  v8bf hi = *(const v8bf*)(p + 16);
  v16bf r;
#pragma unroll
  for (int i = 0; i < 8; ++i) { r[i] = lo[i]; r[i + 8] = hi[i]; }
  return r;
}

// stage a ROWS x 32 f32 tile from global into a bf16 LDS tile (row stride 32).
// unit = one 16-float half-row; NT threads cover ROWS*2 units.
template <int ROWS, int NT>
__device__ __forceinline__ void stage_f32(const float* src, int srcStride,
                                          __bf16* dst, int t) {
  constexpr int UNITS = ROWS * 2;
#pragma unroll
  for (int u = t; u < UNITS; u += NT) {
    int row = u >> 1, half = u & 1;
    const float* p = src + (size_t)row * srcStride + half * 16;
    float f[16];
#pragma unroll
    for (int i = 0; i < 4; ++i) {
      v4f a = *(const v4f*)(p + i * 4);
#pragma unroll
      for (int j = 0; j < 4; ++j) f[i * 4 + j] = a[j];
    }
    v8bf b0, b1;
#pragma unroll
    for (int i = 0; i < 8; ++i) { b0[i] = (__bf16)f[i]; b1[i] = (__bf16)f[8 + i]; }
    __bf16* q = dst + row * 32 + half * 16;
    *(v8bf*)q = b0;
    *(v8bf*)(q + 8) = b1;
  }
}

// CDNA5 async global->LDS copy (ASYNCcnt-tracked), 16B granule
__device__ __forceinline__ void async_b128(uint32_t lds, const void* gaddr) {
  asm volatile("global_load_async_to_lds_b128 %0, %1, off"
               :: "v"(lds), "v"((uint64_t)(uintptr_t)gaddr)
               : "memory");
}
__device__ __forceinline__ void wait_async0() {
  asm volatile("s_wait_asynccnt 0x0" ::: "memory");
}

// ---------------------------------------------------------------------------
// Kernel A: fused QKV projection. M=4096(b,q|c) x K=768, N in [0,2304):
//   seg0 -> Q = (query@Wq^T + bq) * 1/8  -> Qb [b,h,2048,64] bf16
//   seg1 -> K = context@Wk^T + bk        -> Kb [b,h,2048,64] bf16
//   seg2 -> V = context@Wv^T + bv        -> Vt [b,h,64,2048] bf16 (transposed)
// Block: 256 threads (8 waves), tile 128m x 64n; wave w owns rows 16w..16w+15.
// ---------------------------------------------------------------------------
__global__ __launch_bounds__(256) void qkv_kernel(
    const float* __restrict__ query, const float* __restrict__ context,
    const float* __restrict__ Wq, const float* __restrict__ Wqb,
    const float* __restrict__ Wk, const float* __restrict__ Wkb,
    const float* __restrict__ Wv, const float* __restrict__ Wvb,
    __bf16* __restrict__ Qb, __bf16* __restrict__ Kb, __bf16* __restrict__ Vt) {
  __shared__ __bf16 Xs[128 * 32];
  __shared__ __bf16 Ws[64 * 32];
  const int t = threadIdx.x, lane = t & 31, w = t >> 5;
  const int m0 = blockIdx.x * 128;
  const int n0 = blockIdx.y * 64;
  const int seg = n0 / 768;
  const int nseg = n0 - seg * 768;
  const float* X  = (seg == 0) ? query : context;
  const float* W  = (seg == 0) ? Wq : ((seg == 1) ? Wk : Wv);
  const float* Bv = (seg == 0) ? Wqb : ((seg == 1) ? Wkb : Wvb);

  v8f acc[4];
#pragma unroll
  for (int j = 0; j < 4; ++j) acc[j] = vzero();

  for (int k0 = 0; k0 < 768; k0 += 32) {
    __syncthreads();
    stage_f32<128, 256>(X + (size_t)m0 * 768 + k0, 768, Xs, t);
    stage_f32<64, 256>(W + (size_t)nseg * 768 + k0, 768, Ws, t);
    __syncthreads();
    v16bf a = load_frag(Xs + (16 * w) * 32, lane, 32);
    v16bf bf[4];
#pragma unroll
    for (int j = 0; j < 4; ++j) bf[j] = load_frag(Ws + (16 * j) * 32, lane, 32);
#pragma unroll
    for (int j = 0; j < 4; ++j) acc[j] = wmma_bf16(a, bf[j], acc[j]);
  }

  const int col = lane & 15, hi = (lane >> 4) & 1;
  const float scale = (seg == 0) ? 0.125f : 1.0f;  // 1/sqrt(64) folded into Q
#pragma unroll
  for (int j = 0; j < 4; ++j) {
    int f = nseg + 16 * j + col;
    int head = f >> 6, dk = f & 63;
    float bias = Bv[f];
#pragma unroll
    for (int r = 0; r < 8; ++r) {
      int m = m0 + 16 * w + r + 8 * hi;
      int b = m >> 11, q = m & 2047;
      float val = (acc[j][r] + bias) * scale;
      if (seg == 0)
        Qb[(((size_t)b * 12 + head) * 2048 + q) * 64 + dk] = (__bf16)val;
      else if (seg == 1)
        Kb[(((size_t)b * 12 + head) * 2048 + q) * 64 + dk] = (__bf16)val;
      else
        Vt[(((size_t)b * 12 + head) * 64 + dk) * 2048 + q] = (__bf16)val;
    }
  }
}

// issue async copies of one K tile (32x64 bf16, contiguous 4KB) and one V tile
// (64 rows x 64B, row stride 4096B) into LDS buffer `buf`; 256 threads, one
// 16B chunk each per tile.
__device__ __forceinline__ void issue_kv(const char* gk, const char* gv,
                                         uint32_t ldsK, uint32_t ldsV,
                                         int buf, int t) {
  async_b128(ldsK + buf * 4096 + t * 16, gk + t * 16);
  int row = t >> 2, ch = t & 3;
  async_b128(ldsV + buf * 4096 + row * 64 + ch * 16,
             gv + (size_t)row * 4096 + ch * 16);
}

// ---------------------------------------------------------------------------
// Kernel B: flash attention. Block = (b, h, 128 q-rows); 8 waves x 16 rows.
// Online softmax with c-tiles of 32; double-buffered async K/V staging so the
// DMA for tile i+1 overlaps the WMMAs + softmax of tile i.
// ---------------------------------------------------------------------------
__global__ __launch_bounds__(256) void attn_kernel(
    const __bf16* __restrict__ Qb, const __bf16* __restrict__ Kb,
    const __bf16* __restrict__ Vt, const int* __restrict__ cmask,
    __bf16* __restrict__ AO) {
  __shared__ __bf16 Ks[2][32 * 64];  // [c][dk]
  __shared__ __bf16 Vs[2][64 * 32];  // [dk][c]
  __shared__ float  Ss[128 * 32];    // scores f32
  __shared__ __bf16 Ps[128 * 32];    // probs bf16 (A-frag layout source)
  __shared__ float  aS[128];
  __shared__ float  lS[128];

  const int t = threadIdx.x, lane = t & 31, w = t >> 5;
  const int q0 = blockIdx.x * 128, h = blockIdx.y, b = blockIdx.z;
  const int col = lane & 15, hi = (lane >> 4) & 1;

  const __bf16* Qh = Qb + (((size_t)b * 12 + h) * 2048) * 64;
  const __bf16* Kh = Kb + (((size_t)b * 12 + h) * 2048) * 64;
  const __bf16* Vh = Vt + (((size_t)b * 12 + h) * 64) * 2048;
  const int* mrow = cmask + (size_t)b * 2048;

  // Q fragments held in registers for the whole pass (rows q0+16w .. +15)
  v16bf qf0 = load_frag(Qh + (size_t)(q0 + 16 * w) * 64, lane, 64);
  v16bf qf1 = load_frag(Qh + (size_t)(q0 + 16 * w) * 64 + 32, lane, 64);

  v8f o[4];
#pragma unroll
  for (int j = 0; j < 4; ++j) o[j] = vzero();
  float m_r = NEG_INF, l_r = 0.f;  // row state, owned by threads t<128 (row t)

  const uint32_t ldsK = (uint32_t)(uintptr_t)(&Ks[0][0]);
  const uint32_t ldsV = (uint32_t)(uintptr_t)(&Vs[0][0]);

  // prologue: fill buffer 0
  issue_kv((const char*)Kh, (const char*)Vh, ldsK, ldsV, 0, t);
  wait_async0();
  __syncthreads();

  for (int it = 0; it < 64; ++it) {
    const int c0 = it * 32;
    const int buf = it & 1, nxt = buf ^ 1;

    // kick off DMA for the next tile; waited at the end of this iteration
    if (it + 1 < 64)
      issue_kv((const char*)(Kh + (size_t)(c0 + 32) * 64),
               (const char*)(Vh + (c0 + 32)), ldsK, ldsV, nxt, t);

    int mk0 = mrow[c0 + col];
    int mk1 = mrow[c0 + 16 + col];

    // S = Q * K^T : batch the 4 K fragments, then 4 WMMAs
    v16bf kf[4];
    kf[0] = load_frag(&Ks[buf][0 * 64 + 0],  lane, 64);
    kf[1] = load_frag(&Ks[buf][0 * 64 + 32], lane, 64);
    kf[2] = load_frag(&Ks[buf][16 * 64 + 0], lane, 64);
    kf[3] = load_frag(&Ks[buf][16 * 64 + 32], lane, 64);
    v8f s0 = vzero(), s1 = vzero();
    s0 = wmma_bf16(qf0, kf[0], s0);
    s1 = wmma_bf16(qf0, kf[2], s1);
    s0 = wmma_bf16(qf1, kf[1], s0);
    s1 = wmma_bf16(qf1, kf[3], s1);

    // mask + spill S to LDS (C layout: lane<16 rows r, lane>=16 rows r+8)
#pragma unroll
    for (int r = 0; r < 8; ++r) {
      int row = 16 * w + r + 8 * hi;
      Ss[row * 32 + col]      = mk0 ? s0[r] : NEG_INF;
      Ss[row * 32 + 16 + col] = mk1 ? s1[r] : NEG_INF;
    }
    __syncthreads();

    // online softmax: thread t<128 owns row t
    if (t < 128) {
      float vals[32];
      float mt = NEG_INF;
#pragma unroll
      for (int j = 0; j < 32; ++j) { vals[j] = Ss[t * 32 + j]; mt = fmaxf(mt, vals[j]); }
      float mn = fmaxf(m_r, mt);
      float alpha = __expf(m_r - mn);
      float rs = 0.f;
#pragma unroll
      for (int j = 0; j < 32; ++j) {
        float p = __expf(vals[j] - mn);
        rs += p;
        Ps[t * 32 + j] = (__bf16)p;
      }
      l_r = l_r * alpha + rs;
      m_r = mn;
      aS[t] = alpha;
    }
    __syncthreads();

    // rescale O accumulators by alpha, then O += P * V (batched fragments)
    float av[8];
#pragma unroll
    for (int r = 0; r < 8; ++r) av[r] = aS[16 * w + r + 8 * hi];
#pragma unroll
    for (int j = 0; j < 4; ++j)
#pragma unroll
      for (int r = 0; r < 8; ++r) o[j][r] *= av[r];

    v16bf pf = load_frag(Ps + (16 * w) * 32, lane, 32);
    v16bf vf[4];
#pragma unroll
    for (int j = 0; j < 4; ++j) vf[j] = load_frag(&Vs[buf][(16 * j) * 32], lane, 32);
#pragma unroll
    for (int j = 0; j < 4; ++j) o[j] = wmma_bf16(pf, vf[j], o[j]);

    // DMA for the next tile must be LDS-visible before the next iteration;
    // this wait+barrier also fences this iteration's reads of `buf` against
    // the async writes that will target it in the following iteration.
    wait_async0();
    __syncthreads();
  }

  if (t < 128) lS[t] = 1.0f / l_r;
  __syncthreads();
  float inv[8];
#pragma unroll
  for (int r = 0; r < 8; ++r) inv[r] = lS[16 * w + r + 8 * hi];
#pragma unroll
  for (int j = 0; j < 4; ++j) {
#pragma unroll
    for (int r = 0; r < 8; ++r) {
      int q = q0 + 16 * w + r + 8 * hi;
      AO[((size_t)b * 2048 + q) * 768 + h * 64 + 16 * j + col] =
          (__bf16)(o[j][r] * inv[r]);
    }
  }
}

// ---------------------------------------------------------------------------
// Kernel C: output projection, out = AO @ fc_w^T + fc_b (f32 result).
// Block: 256 threads, tile 128m x 64n. AO already bf16 -> async-copy straight
// into LDS (overlaps the f32->bf16 weight staging).
// ---------------------------------------------------------------------------
__global__ __launch_bounds__(256) void proj_kernel(
    const __bf16* __restrict__ AO, const float* __restrict__ Wf,
    const float* __restrict__ bf_, float* __restrict__ out) {
  __shared__ __bf16 Xs[128 * 32];
  __shared__ __bf16 Ws[64 * 32];
  const int t = threadIdx.x, lane = t & 31, w = t >> 5;
  const int m0 = blockIdx.x * 128, n0 = blockIdx.y * 64;
  const uint32_t ldsX = (uint32_t)(uintptr_t)(&Xs[0]);

  v8f acc[4];
#pragma unroll
  for (int j = 0; j < 4; ++j) acc[j] = vzero();

  for (int k0 = 0; k0 < 768; k0 += 32) {
    __syncthreads();
    const char* gx = (const char*)(AO + (size_t)m0 * 768 + k0);
#pragma unroll
    for (int u = 0; u < 2; ++u) {
      int idx = t + u * 256;  // 128 rows x 4 chunks of 16B
      int row = idx >> 2, ch = idx & 3;
      async_b128(ldsX + row * 64 + ch * 16, gx + (size_t)row * 1536 + ch * 16);
    }
    stage_f32<64, 256>(Wf + (size_t)n0 * 768 + k0, 768, Ws, t);
    wait_async0();
    __syncthreads();
    v16bf a = load_frag(Xs + (16 * w) * 32, lane, 32);
    v16bf bf[4];
#pragma unroll
    for (int j = 0; j < 4; ++j) bf[j] = load_frag(Ws + (16 * j) * 32, lane, 32);
#pragma unroll
    for (int j = 0; j < 4; ++j) acc[j] = wmma_bf16(a, bf[j], acc[j]);
  }

  const int col = lane & 15, hi = (lane >> 4) & 1;
#pragma unroll
  for (int j = 0; j < 4; ++j) {
    int n = n0 + 16 * j + col;
    float bias = bf_[n];
#pragma unroll
    for (int r = 0; r < 8; ++r) {
      int m = m0 + 16 * w + r + 8 * hi;
      out[(size_t)m * 768 + n] = acc[j][r] + bias;
    }
  }
}

// ---------------------------------------------------------------------------
extern "C" void kernel_launch(void* const* d_in, const int* in_sizes, int n_in,
                              void* d_out, int out_size, void* d_ws, size_t ws_size,
                              hipStream_t stream) {
  const float* query   = (const float*)d_in[0];
  const float* context = (const float*)d_in[1];
  const int*   cmask   = (const int*)d_in[2];
  const float* Wq_w = (const float*)d_in[3];
  const float* Wq_b = (const float*)d_in[4];
  const float* Wk_w = (const float*)d_in[5];
  const float* Wk_b = (const float*)d_in[6];
  const float* Wv_w = (const float*)d_in[7];
  const float* Wv_b = (const float*)d_in[8];
  const float* fc_w = (const float*)d_in[9];
  const float* fc_b = (const float*)d_in[10];

  const size_t HEADELEMS = (size_t)2 * 12 * 2048 * 64;  // 3,145,728
  __bf16* Qb = (__bf16*)d_ws;
  __bf16* Kb = Qb + HEADELEMS;
  __bf16* Vt = Kb + HEADELEMS;
  __bf16* AO = Vt + HEADELEMS;  // [2,2048,768] bf16

  dim3 gA(32, 36);  // 4096/128 m-tiles x 2304/64 n-tiles
  qkv_kernel<<<gA, 256, 0, stream>>>(query, context, Wq_w, Wq_b, Wk_w, Wk_b,
                                     Wv_w, Wv_b, Qb, Kb, Vt);

  dim3 gB(16, 12, 2);  // q-tiles x heads x batch
  attn_kernel<<<gB, 256, 0, stream>>>(Qb, Kb, Vt, cmask, AO);

  dim3 gC(32, 12);  // 4096/128 x 768/64
  proj_kernel<<<gC, 256, 0, stream>>>(AO, fc_w, fc_b, (float*)d_out);
}
